// DiTAttention_8761733284132
// MI455X (gfx1250) — matrix operand order repaired and checked by reading, not compiled, not run on hardware
//
#include <hip/hip_runtime.h>
#include <hip/hip_bf16.h>
#include <stdint.h>

// ---------------------------------------------------------------------------
// DiT attention for MI455X (gfx1250, wave32, WMMA, async-to-LDS staging).
// fp32 in/out, f16 WMMA compute with fp32 accumulation.
// ---------------------------------------------------------------------------

#define HDIM 1536
#define NH   24
#define HD   64
#define BB   2
#define SS   2048
#define MTOT (BB * SS)      // 4096 rows

typedef _Float16 v16h __attribute__((ext_vector_type(16)));
typedef _Float16 v8h  __attribute__((ext_vector_type(8)));
typedef _Float16 v4h  __attribute__((ext_vector_type(4)));
typedef float    v8f  __attribute__((ext_vector_type(8)));

// ---- CDNA5 async global->LDS copy (ASYNCcnt-tracked; keeps compiler's
// DScnt/LOADcnt bookkeeping untouched) ------------------------------------
__device__ __forceinline__ void async_b128(uint32_t lds_addr, uint64_t gaddr) {
  asm volatile("global_load_async_to_lds_b128 %0, %1, off"
               :: "v"(lds_addr), "v"(gaddr) : "memory");
}
__device__ __forceinline__ void wait_async0() {
  asm volatile("s_wait_asynccnt 0x0" ::: "memory");
}
__device__ __forceinline__ void wait_async4() {
  asm volatile("s_wait_asynccnt 0x4" ::: "memory");
}
__device__ __forceinline__ uint32_t lds_addr_of(const void* p) {
  return (uint32_t)(uintptr_t)p;   // low 32 bits of flat shared addr = LDS offset
}
__device__ __forceinline__ v16h cat8(v8h a, v8h b) {
  v16h r;
#pragma unroll
  for (int i = 0; i < 8; ++i) { r[i] = a[i]; r[i + 8] = b[i]; }
  return r;
}

// ---------------------------------------------------------------------------
// fp32 -> f16 conversion (4 elements / thread)
// ---------------------------------------------------------------------------
__global__ void cvt_f32_f16_kernel(const float* __restrict__ src,
                                   _Float16* __restrict__ dst, int n4) {
  int i = blockIdx.x * blockDim.x + threadIdx.x;
  if (i < n4) {
    const float4 v = ((const float4*)src)[i];
    v4h o;
    o[0] = (_Float16)v.x; o[1] = (_Float16)v.y;
    o[2] = (_Float16)v.z; o[3] = (_Float16)v.w;
    *(v4h*)(dst + (size_t)i * 4) = o;
  }
}

// ---------------------------------------------------------------------------
// GEMM (NT): C[m,n] = sum_k A[m,k] * W[n,k] + bias[n]
//   A: [MTOT, HDIM] f16 row-major, W: [HDIM, HDIM] f16 row-major.
// MODE 0: store f16 into per-head layout [B, NH, S, HD]
// MODE 1: store fp32 row-major [MTOT, HDIM]
// Workgroup tile 128x128, 8 waves of 32x64, K-step 32.
// Double-buffered LDS staging via global_load_async_to_lds_b128.
// ---------------------------------------------------------------------------
#define GP 40   // LDS pitch in halves (80 B; rows 16-B aligned)

template <int MODE>
__global__ __launch_bounds__(256) void gemm_nt_kernel(
    const _Float16* __restrict__ A, const _Float16* __restrict__ W,
    const float* __restrict__ bias, void* __restrict__ dstv) {
  __shared__ _Float16 Asl[2][128 * GP];
  __shared__ _Float16 Bsl[2][128 * GP];

  const int tid  = threadIdx.x;
  const int m0   = blockIdx.y * 128;
  const int n0   = blockIdx.x * 128;
  const int w    = tid >> 5;
  const int lane = tid & 31;
  const int half = lane >> 4;
  const int r16  = lane & 15;
  const int wm   = (w & 3) * 32;   // 4 waves along M
  const int wn   = (w >> 2) * 64;  // 2 waves along N

  v8f acc[2][4];
#pragma unroll
  for (int fm = 0; fm < 2; ++fm)
#pragma unroll
    for (int nf = 0; nf < 4; ++nf)
#pragma unroll
      for (int r = 0; r < 8; ++r) acc[fm][nf][r] = 0.0f;

  const int ldrow = tid >> 1;         // 0..127
  const int ldcol = (tid & 1) * 16;   // 0 or 16

  auto issue_tile = [&](int buf, int k0) {
    const uint64_t ga = (uint64_t)(A + (size_t)(m0 + ldrow) * HDIM + k0 + ldcol);
    const uint64_t gb = (uint64_t)(W + (size_t)(n0 + ldrow) * HDIM + k0 + ldcol);
    const uint32_t la = lds_addr_of(&Asl[buf][ldrow * GP + ldcol]);
    const uint32_t lb = lds_addr_of(&Bsl[buf][ldrow * GP + ldcol]);
    async_b128(la,      ga);
    async_b128(la + 16, ga + 16);
    async_b128(lb,      gb);
    async_b128(lb + 16, gb + 16);
  };

  issue_tile(0, 0);

  const int NKK = HDIM / 32;  // 48
  for (int kk = 0; kk < NKK; ++kk) {
    if (kk + 1 < NKK) { issue_tile((kk + 1) & 1, (kk + 1) * 32); wait_async4(); }
    else             { wait_async0(); }
    __syncthreads();

    const _Float16* As = Asl[kk & 1];
    const _Float16* Bs = Bsl[kk & 1];

    // ---- A fragments (16x32 f16; two contiguous 16-B runs per lane) ----
    v16h af[2];
#pragma unroll
    for (int fm = 0; fm < 2; ++fm) {
      const int row = wm + fm * 16 + r16;
      const v8h a0 = *(const v8h*)&As[row * GP + 8 * half];
      const v8h a1 = *(const v8h*)&As[row * GP + 16 + 8 * half];
      af[fm] = cat8(a0, a1);
    }

    // ---- B fragments (16x32 f16; two contiguous 16-B runs per lane) + WMMA ----
#pragma unroll
    for (int nf = 0; nf < 4; ++nf) {
      const int col = wn + nf * 16 + r16;
      const v8h b0 = *(const v8h*)&Bs[col * GP + 8 * half];
      const v8h b1 = *(const v8h*)&Bs[col * GP + 16 + 8 * half];
      const v16h bf = cat8(b0, b1);
#pragma unroll
      for (int fm = 0; fm < 2; ++fm)
        acc[fm][nf] = __builtin_amdgcn_wmma_f32_16x16x32_f16(
            false, af[fm], false, bf, (short)0, acc[fm][nf], false, false);
    }
    __syncthreads();
  }

  // ---- epilogue ----
#pragma unroll
  for (int fm = 0; fm < 2; ++fm)
#pragma unroll
    for (int nf = 0; nf < 4; ++nf)
#pragma unroll
      for (int r = 0; r < 8; ++r) {
        const int m = m0 + wm + fm * 16 + r + 8 * half;
        const int n = n0 + wn + nf * 16 + r16;
        const float v = acc[fm][nf][r] + bias[n];
        if (MODE == 0) {
          _Float16* dst = (_Float16*)dstv;       // [B, NH, S, HD] f16
          const int b = m >> 11;                 // / SS
          const int s = m & (SS - 1);
          const int h = n >> 6;
          const int d = n & 63;
          dst[(((size_t)(b * NH + h)) * SS + s) * HD + d] = (_Float16)v;
        } else {
          ((float*)dstv)[(size_t)m * HDIM + n] = v;
        }
      }
}

// ---------------------------------------------------------------------------
// RoPE in place on Q and K (per-head layout, f16), fp32 math.
// Q is additionally scaled by 1/sqrt(HD) = 0.125 (folded softmax scale).
// ---------------------------------------------------------------------------
__global__ __launch_bounds__(256) void rope_kernel(
    _Float16* __restrict__ Qh, _Float16* __restrict__ Kh,
    const float* __restrict__ cosb, const float* __restrict__ sinb) {
  const int i  = blockIdx.x * blockDim.x + threadIdx.x;  // < B*NH*S*32
  const int j  = i & 31;
  const int s  = (i >> 5) & (SS - 1);
  const int bh = i >> 16;  // 32*2048 elements per (b,h)
  const size_t base = ((size_t)bh * SS + s) * HD;
  const float c   = cosb[s * HD + j];
  const float sn0 = sinb[s * HD + j];
  const float sn1 = sinb[s * HD + j + 32];

  const float q0 = (float)Qh[base + j];
  const float q1 = (float)Qh[base + j + 32];
  Qh[base + j]      = (_Float16)((q0 * c - q1 * sn0) * 0.125f);
  Qh[base + j + 32] = (_Float16)((q1 * c + q0 * sn1) * 0.125f);

  const float k0 = (float)Kh[base + j];
  const float k1 = (float)Kh[base + j + 32];
  Kh[base + j]      = (_Float16)(k0 * c - k1 * sn0);
  Kh[base + j + 32] = (_Float16)(k1 * c + k0 * sn1);
}

// ---------------------------------------------------------------------------
// Flash attention, transposed formulation:
//   S^T = K (keys x d)  .  Q^T (d x q)      -> each lane owns ONE query column
//   O^T = V^T (d x keys) . P^T (keys x q)
// Softmax state (m, l) is per-lane scalar; only one shfl_xor(16) per reduction.
// P^T never touches LDS: lane<->lane+16 exchange of packed f16 pairs.
// Grid (S/128, B*NH), 256 threads (8 waves x 16 query rows).
// ---------------------------------------------------------------------------
#define KVP 72  // LDS pitch in halves (144 B; rows 16-B aligned)

__global__ __launch_bounds__(256) void attn_kernel(
    const _Float16* __restrict__ Q, const _Float16* __restrict__ K,
    const _Float16* __restrict__ V, _Float16* __restrict__ Out) {
  __shared__ _Float16 Kt[64 * KVP];        // [key][d]   row-major
  __shared__ _Float16 VT[64 * KVP];        // [d][key]   V transposed
  __shared__ _Float16 Ot[8 * 16 * KVP];    // per-wave [q_local][d] transpose buf

  const int bh   = blockIdx.y;        // 0..47
  const int b    = bh / NH;
  const int h    = bh % NH;
  const int q0   = blockIdx.x * 128;
  const int tid  = threadIdx.x;
  const int w    = tid >> 5;
  const int lane = tid & 31;
  const int half = lane >> 4;
  const int r16  = lane & 15;

  const size_t head_base = (size_t)bh * SS * HD;
  const _Float16* Qh = Q + head_base;
  const _Float16* Kh = K + head_base;
  const _Float16* Vh = V + head_base;

  // ---- Q^T B-fragments: lane owns query column q = qr0 + r16 -------------
  const int qr0 = q0 + w * 16;
  const int qcol = qr0 + r16;
  v16h qb[2];
#pragma unroll
  for (int ks = 0; ks < 2; ++ks) {
    const _Float16* p = Qh + (size_t)qcol * HD + 8 * half + 32 * ks;
    qb[ks] = cat8(*(const v8h*)p, *(const v8h*)(p + 16));
  }

  v8f o[4];                 // O^T: 4 d-blocks x (8 d-rows) for column qcol
#pragma unroll
  for (int nf = 0; nf < 4; ++nf)
#pragma unroll
    for (int r = 0; r < 8; ++r) o[nf][r] = 0.0f;
  float m_i = -3.0e38f, l_i = 0.0f;

  // staging maps
  const int krow = tid >> 2;        // 0..63  (K tile rows)
  const int kcol = (tid & 3) * 16;  // 0,16,32,48
  const int kp   = tid >> 3;        // 0..31  (V key pair)
  const int dc   = tid & 7;         // 0..7   (V d-chunk of 8)

  union HU { _Float16 h[2]; int u; };

  for (int kv0 = 0; kv0 < SS; kv0 += 64) {
    // ---- stage K tile via async DMA to LDS ----
    {
      const uint64_t gk = (uint64_t)(Kh + (size_t)(kv0 + krow) * HD + kcol);
      const uint32_t lk = lds_addr_of(&Kt[krow * KVP + kcol]);
      async_b128(lk, gk);
      async_b128(lk + 16, gk + 16);
    }
    // ---- stage V transposed: pack key pairs, store b32 ----
    {
      const v8h va = *(const v8h*)(Vh + (size_t)(kv0 + 2 * kp) * HD + dc * 8);
      const v8h vb = *(const v8h*)(Vh + (size_t)(kv0 + 2 * kp + 1) * HD + dc * 8);
#pragma unroll
      for (int j = 0; j < 8; ++j) {
        HU t; t.h[0] = va[j]; t.h[1] = vb[j];
        *(int*)&VT[(dc * 8 + j) * KVP + 2 * kp] = t.u;
      }
    }
    wait_async0();
    __syncthreads();

    // ---- S^T = K . Q^T : 4 key-blocks of 16 ----
    v8f st[4];
#pragma unroll
    for (int kb = 0; kb < 4; ++kb) {
      v8f s;
#pragma unroll
      for (int r = 0; r < 8; ++r) s[r] = 0.0f;
#pragma unroll
      for (int ks = 0; ks < 2; ++ks) {
        const _Float16* kr = &Kt[(kb * 16 + r16) * KVP + 32 * ks];
        const v16h ka = cat8(*(const v8h*)(kr + 8 * half),
                             *(const v8h*)(kr + 16 + 8 * half));
        s = __builtin_amdgcn_wmma_f32_16x16x32_f16(false, ka, false, qb[ks],
                                                   (short)0, s, false, false);
      }
      st[kb] = s;
    }

    // ---- per-lane online softmax for query column qcol ----
    float mn = m_i;
#pragma unroll
    for (int kb = 0; kb < 4; ++kb)
#pragma unroll
      for (int r = 0; r < 8; ++r) mn = fmaxf(mn, st[kb][r]);
    mn = fmaxf(mn, __shfl_xor(mn, 16, 32));

    const float alpha = __expf(m_i - mn);
    m_i = mn;
    float rsum = 0.0f;
#pragma unroll
    for (int kb = 0; kb < 4; ++kb)
#pragma unroll
      for (int r = 0; r < 8; ++r) {
        const float p = __expf(st[kb][r] - mn);
        st[kb][r] = p;
        rsum += p;
      }
    rsum += __shfl_xor(rsum, 16, 32);
    l_i = l_i * alpha + rsum;
#pragma unroll
    for (int nf = 0; nf < 4; ++nf)
#pragma unroll
      for (int r = 0; r < 8; ++r) o[nf][r] *= alpha;

    // ---- build P^T B-fragments: pack f16 pairs, exchange lane<->lane+16 ----
    int pk[4][4];
#pragma unroll
    for (int kb = 0; kb < 4; ++kb)
#pragma unroll
      for (int j = 0; j < 4; ++j) {
        HU t;
        t.h[0] = (_Float16)st[kb][2 * j];
        t.h[1] = (_Float16)st[kb][2 * j + 1];
        pk[kb][j] = t.u;
      }

    v16h pb[2];
#pragma unroll
    for (int ks = 0; ks < 2; ++ks) {
      union VU { int u[8]; v16h h; } vu;
#pragma unroll
      for (int j = 0; j < 4; ++j) {
        const int own = half ? pk[2 * ks + 1][j] : pk[2 * ks][j];
        const int sel = half ? pk[2 * ks][j]     : pk[2 * ks + 1][j];
        const int oth = __shfl_xor(sel, 16, 32);
        // half0: K 0..7 own(block 2ks), K 8..15 partner(block 2ks)
        // half1: K 16..23 partner(block 2ks+1), K 24..31 own(block 2ks+1)
        vu.u[j]     = half ? oth : own;
        vu.u[j + 4] = half ? own : oth;
      }
      pb[ks] = vu.h;
    }

    // ---- O^T += V^T . P^T ----
#pragma unroll
    for (int nf = 0; nf < 4; ++nf) {
#pragma unroll
      for (int ks = 0; ks < 2; ++ks) {
        const _Float16* vr = &VT[(nf * 16 + r16) * KVP + 32 * ks];
        const v16h va = cat8(*(const v8h*)(vr + 16 * half),
                             *(const v8h*)(vr + 16 * half + 8));
        o[nf] = __builtin_amdgcn_wmma_f32_16x16x32_f16(false, va, false, pb[ks],
                                                       (short)0, o[nf], false, false);
      }
    }
    __syncthreads();
  }

  // ---- normalize, transpose back through per-wave LDS, store b128 --------
  const float invl = 1.0f / l_i;
  _Float16* Ow = &Ot[w * 16 * KVP];
#pragma unroll
  for (int nf = 0; nf < 4; ++nf) {
    v8h row;
#pragma unroll
    for (int r = 0; r < 8; ++r) row[r] = (_Float16)(o[nf][r] * invl);
    // lane's 8 d-values are contiguous: d = nf*16 + 8*half + r
    *(v8h*)&Ow[r16 * KVP + nf * 16 + 8 * half] = row;
  }
  // LDS in-order within wave: rows now complete; lane reads row r16, half*32..
#pragma unroll
  for (int j = 0; j < 2; ++j) {
    const v8h d0 = *(const v8h*)&Ow[r16 * KVP + 32 * half + 16 * j];
    const v8h d1 = *(const v8h*)&Ow[r16 * KVP + 32 * half + 16 * j + 8];
    _Float16* dst = Out + ((size_t)b * SS + qr0 + r16) * HDIM + h * HD +
                    32 * half + 16 * j;
    *(v8h*)dst = d0;
    *(v8h*)(dst + 8) = d1;
  }
}

// ---------------------------------------------------------------------------
// Host-side launcher
// ---------------------------------------------------------------------------
extern "C" void kernel_launch(void* const* d_in, const int* in_sizes, int n_in,
                              void* d_out, int out_size, void* d_ws, size_t ws_size,
                              hipStream_t stream) {
  (void)in_sizes; (void)n_in; (void)out_size; (void)ws_size;

  const float* X    = (const float*)d_in[0];
  const float* rc   = (const float*)d_in[1];
  const float* rs   = (const float*)d_in[2];
  const float* wq   = (const float*)d_in[3];
  const float* bq   = (const float*)d_in[4];
  const float* wk   = (const float*)d_in[5];
  const float* bk   = (const float*)d_in[6];
  const float* wv   = (const float*)d_in[7];
  const float* bv   = (const float*)d_in[8];
  const float* wo   = (const float*)d_in[9];
  const float* bo   = (const float*)d_in[10];
  float* out        = (float*)d_out;

  // workspace layout (all f16; every segment 16-B aligned)
  const size_t nX = (size_t)MTOT * HDIM;      // 6,291,456
  const size_t nW = (size_t)HDIM * HDIM;      // 2,359,296
  _Float16* Xh  = (_Float16*)d_ws;
  _Float16* Wqh = Xh  + nX;
  _Float16* Wkh = Wqh + nW;
  _Float16* Wvh = Wkh + nW;
  _Float16* Woh = Wvh + nW;
  _Float16* Qh  = Woh + nW;   // [B, NH, S, HD]
  _Float16* Kh  = Qh  + nX;
  _Float16* Vh  = Kh  + nX;
  _Float16* AOh = Vh  + nX;   // [B, S, H]

  // 1) convert inputs to f16
  {
    const int t = 256;
    cvt_f32_f16_kernel<<<(int)(nX / 4 / t), t, 0, stream>>>(X,  Xh,  (int)(nX / 4));
    cvt_f32_f16_kernel<<<(int)(nW / 4 / t), t, 0, stream>>>(wq, Wqh, (int)(nW / 4));
    cvt_f32_f16_kernel<<<(int)(nW / 4 / t), t, 0, stream>>>(wk, Wkh, (int)(nW / 4));
    cvt_f32_f16_kernel<<<(int)(nW / 4 / t), t, 0, stream>>>(wv, Wvh, (int)(nW / 4));
    cvt_f32_f16_kernel<<<(int)(nW / 4 / t), t, 0, stream>>>(wo, Woh, (int)(nW / 4));
  }

  // 2) QKV projections (per-head f16 output layout)
  {
    dim3 g(HDIM / 128, MTOT / 128);  // (12, 32)
    gemm_nt_kernel<0><<<g, 256, 0, stream>>>(Xh, Wqh, bq, (void*)Qh);
    gemm_nt_kernel<0><<<g, 256, 0, stream>>>(Xh, Wkh, bk, (void*)Kh);
    gemm_nt_kernel<0><<<g, 256, 0, stream>>>(Xh, Wvh, bv, (void*)Vh);
  }

  // 3) RoPE on Q, K (Q also picks up the 1/sqrt(HD) scale)
  {
    const int total = BB * NH * SS * 32;  // 3,145,728
    rope_kernel<<<total / 256, 256, 0, stream>>>(Qh, Kh, rc, rs);
  }

  // 4) attention (transposed flash formulation)
  {
    dim3 g(SS / 128, BB * NH);  // (16, 48)
    attn_kernel<<<g, 256, 0, stream>>>(Qh, Kh, Vh, AOh);
  }

  // 5) output projection -> fp32 d_out
  {
    dim3 g(HDIM / 128, MTOT / 128);
    gemm_nt_kernel<1><<<g, 256, 0, stream>>>(AOh, Woh, bo, (void*)out);
  }
}
